// VectorQuantizer_43808666419909
// MI455X (gfx1250) — matrix operand order repaired and checked by reading, not compiled, not run on hardware
//
#include <hip/hip_runtime.h>
#include <hip/hip_bf16.h>
#include <stdint.h>

// ---------------------------------------------------------------------------
// VQ-VAE vector quantizer for MI455X (gfx1250, wave32, WMMA).
// Pipeline:
//   0) prep_cb   : cb f32 -> cb bf16, cbnorm[j] = ||cb_j||^2
//   1) outtab    : outTab = cb @ W_out^T + b_out   (f32 WMMA 16x16x4)
//   2) proj_in   : zp = z @ W_in^T + b_in          (f32 WMMA 16x16x4), store f32+bf16
//   3) argmin    : scores = cbnorm - 2*zp@cb^T     (bf16 WMMA 16x16x32, f32 acc),
//                  codebook staged in 264KB LDS via GLOBAL_LOAD_ASYNC_TO_LDS
//   4) gather    : out[p, co] = outTab[idx[p], co] (pure gather - GEMM eliminated)
//   5) loss      : exact f32 (1+beta)*mean((cb[idx]-zp)^2)
// ---------------------------------------------------------------------------

#define C_IN    256
#define C_CODE  64
#define N_CODES 2048
#define C_OUT   256
#define P_TOTAL 65536          // 64 * 32 * 32 pixels
#define BETA    0.25f

typedef __attribute__((ext_vector_type(2)))  float  v2f;
typedef __attribute__((ext_vector_type(8)))  float  v8f;
typedef __attribute__((ext_vector_type(16))) __bf16 v16bf;

union F8      { v8f   v; float f[8]; };
union BF16x16 { v16bf v; uint4 q[2]; };   // 32 bytes: two b128 chunks

// ---------------------------------------------------------------------------
// Kernel 0: codebook prep (bf16 copy + squared norms)
// ---------------------------------------------------------------------------
__global__ void vq_prep_cb(const float* __restrict__ cb,
                           __bf16* __restrict__ cb16,
                           float* __restrict__ cbnorm) {
    int j = blockIdx.x * blockDim.x + threadIdx.x;
    if (j >= N_CODES) return;
    float s = 0.f;
    #pragma unroll 8
    for (int k = 0; k < C_CODE; ++k) {
        float v = cb[j * C_CODE + k];
        s += v * v;
        cb16[j * C_CODE + k] = (__bf16)v;
    }
    cbnorm[j] = s;
}

// ---------------------------------------------------------------------------
// Kernel 1: outTab[j,co] = sum_k cb[j,k]*W_out[co,k] + b_out[co]
//   M=2048 (codes), N=256, K=64 ; f32 WMMA 16x16x4
//   A-frag f32 16x4: lane m=l&15, hi=l>>4, VGPR v holds K = k0 + 2*hi + v
// ---------------------------------------------------------------------------
__global__ void vq_outtab(const float* __restrict__ cb,
                          const float* __restrict__ Wout,
                          const float* __restrict__ bout,
                          float* __restrict__ outtab) {
    int lane = threadIdx.x & 31;
    int wave = threadIdx.x >> 5;
    int jt   = blockIdx.x * 8 + wave;      // 0..127 row tiles
    int m    = lane & 15;
    int hi   = lane >> 4;
    int jrow = jt * 16 + m;

    for (int nt = 0; nt < 16; ++nt) {
        int co = nt * 16 + m;
        F8 acc{};
        #pragma unroll
        for (int k0 = 0; k0 < C_CODE; k0 += 4) {
            int k = k0 + 2 * hi;
            v2f A = { cb[jrow * C_CODE + k], cb[jrow * C_CODE + k + 1] };
            v2f B = { Wout[co * C_CODE + k], Wout[co * C_CODE + k + 1] };
            acc.v = __builtin_amdgcn_wmma_f32_16x16x4_f32(
                false, A, false, B, (short)0, acc.v, false, false);
        }
        float bias = bout[co];
        #pragma unroll
        for (int r = 0; r < 8; ++r) {
            int j = jt * 16 + r + 8 * hi;   // C/D layout: VGPR r, half hi -> row
            outtab[j * C_OUT + co] = acc.f[r] + bias;
        }
    }
}

// ---------------------------------------------------------------------------
// Kernel 2: project_in. zp[p,d] = sum_c z[b,c,hw]*W_in[d,c] + b_in[d]
//   M=65536 pixels (16/wave), N=64 (4 tiles), K=256 ; f32 WMMA 16x16x4
//   W_in (64KB) staged in LDS. Stores zp in f32 (for exact loss) and bf16
//   (for the distance GEMM).
// ---------------------------------------------------------------------------
__global__ void vq_proj_in(const float* __restrict__ z,
                           const float* __restrict__ Win,
                           const float* __restrict__ bin,
                           float* __restrict__ zp32,
                           __bf16* __restrict__ zp16) {
    extern __shared__ char dyn_smem[];
    float* sW = (float*)dyn_smem;                       // 16384 floats = 64 KB
    for (int i = threadIdx.x; i < C_CODE * C_IN / 4; i += blockDim.x)
        ((float4*)sW)[i] = ((const float4*)Win)[i];
    __syncthreads();

    int lane = threadIdx.x & 31;
    int wave = threadIdx.x >> 5;
    int mt   = blockIdx.x * 8 + wave;       // 0..4095 pixel tiles
    int m    = lane & 15;
    int hi   = lane >> 4;
    int p    = mt * 16 + m;
    int b    = p >> 10;
    int hw   = p & 1023;
    const float* zb = z + (size_t)b * C_IN * 1024 + hw;  // [c] stride 1024

    F8 acc[4] = {};
    for (int k0 = 0; k0 < C_IN; k0 += 4) {
        int c = k0 + 2 * hi;
        v2f A = { zb[(size_t)c * 1024], zb[(size_t)(c + 1) * 1024] };
        #pragma unroll
        for (int nt = 0; nt < 4; ++nt) {
            int d = nt * 16 + m;
            v2f B = { sW[d * C_IN + c], sW[d * C_IN + c + 1] };
            acc[nt].v = __builtin_amdgcn_wmma_f32_16x16x4_f32(
                false, A, false, B, (short)0, acc[nt].v, false, false);
        }
    }
    #pragma unroll
    for (int nt = 0; nt < 4; ++nt) {
        int d = nt * 16 + m;
        float bias = bin[d];
        #pragma unroll
        for (int r = 0; r < 8; ++r) {
            int pr = mt * 16 + r + 8 * hi;
            float v = acc[nt].f[r] + bias;
            zp32[(size_t)pr * C_CODE + d] = v;
            zp16[(size_t)pr * C_CODE + d] = (__bf16)v;
        }
    }
}

// ---------------------------------------------------------------------------
// Kernel 3: distances + argmin.
//   score[p,j] = cbnorm[j] - 2 * zp[p,:] . cb[j,:]   (||zp||^2 is row-constant)
//   bf16 WMMA 16x16x32 with f32 accumulate; K=64 -> 2 WMMAs per 16x16 tile.
//   Whole bf16 codebook (256KB) + cbnorm (8KB) staged into the 320KB WGP LDS
//   with GLOBAL_LOAD_ASYNC_TO_LDS_B128 (ASYNCcnt path, no VGPR roundtrip).
//   A-frag (16-bit 16x32): lane m=l&15, hi=l>>4 holds K = k0+8*hi+0..7 (v0..3)
//   and K = k0+16+8*hi+0..7 (v4..7) -> two contiguous b128 chunks.
//   B-frag (16-bit 32x16): lane n=l&15 holds K = k0+8*hi+0..7 and k0+16+8*hi+0..7 -> two b128s.
// ---------------------------------------------------------------------------
__global__ void vq_argmin(const __bf16* __restrict__ zp16,
                          const __bf16* __restrict__ cb16,
                          const float* __restrict__ cbnorm,
                          int* __restrict__ idx) {
    extern __shared__ char dyn_smem[];
    __bf16* scb   = (__bf16*)dyn_smem;                       // 2048*64 bf16
    float*  snorm = (float*)(dyn_smem + N_CODES * C_CODE * 2);

    {   // async stage: 256KB codebook + 8KB norms, global -> LDS direct
        unsigned dynbase = __builtin_amdgcn_groupstaticsize();   // dyn-LDS byte base
        unsigned tid = threadIdx.x;
        const char* gcb = (const char*)cb16;
        #pragma unroll 4
        for (int i = 0; i < (N_CODES * C_CODE * 2) / (256 * 16); ++i) {  // 64
            unsigned off = (tid + i * 256u) * 16u;
            asm volatile("global_load_async_to_lds_b128 %0, %1, off"
                         :: "v"(dynbase + off), "v"(gcb + off) : "memory");
        }
        const char* gnm = (const char*)cbnorm;
        unsigned nbase = dynbase + N_CODES * C_CODE * 2;
        #pragma unroll
        for (int i = 0; i < N_CODES / 256; ++i) {                        // 8
            unsigned off = (tid + i * 256u) * 4u;
            asm volatile("global_load_async_to_lds_b32 %0, %1, off"
                         :: "v"(nbase + off), "v"(gnm + off) : "memory");
        }
        asm volatile("s_wait_asynccnt 0" ::: "memory");
    }
    __syncthreads();

    int lane = threadIdx.x & 31;
    int wave = threadIdx.x >> 5;
    int m    = lane & 15;
    int hi   = lane >> 4;

    for (int t = 0; t < 2; ++t) {
        int tile = (blockIdx.x * 8 + wave) * 2 + t;   // 0..4095
        int p    = tile * 16 + m;
        const uint4* zr = (const uint4*)(zp16 + (size_t)p * C_CODE);
        if (t == 0)  // prefetch next tile's zp rows while this one computes
            __builtin_prefetch(zp16 + (size_t)(p + 16) * C_CODE, 0, 1);
        BF16x16 a0, a1;                 // A reused across all 128 code tiles
        a0.q[0] = zr[hi];               // k0=0 : d =      8*hi .. +7
        a0.q[1] = zr[2 + hi];           //        d = 16 + 8*hi .. +7
        a1.q[0] = zr[4 + hi];           // k0=32
        a1.q[1] = zr[6 + hi];

        float minv[8]; int mini[8];
        #pragma unroll
        for (int r = 0; r < 8; ++r) { minv[r] = 3.4e38f; mini[r] = 0x7fffffff; }

        for (int jt = 0; jt < N_CODES / 16; ++jt) {
            int j = jt * 16 + m;
            const uint4* cr = (const uint4*)(scb + (size_t)j * C_CODE);
            BF16x16 b0, b1;
            b0.q[0] = cr[hi];               // k = 8*hi + 0..7
            b0.q[1] = cr[2 + hi];           // k = 16 + 8*hi + 0..7
            b1.q[0] = cr[4 + hi];           // k0=32
            b1.q[1] = cr[6 + hi];

            F8 acc{};
            acc.v = __builtin_amdgcn_wmma_f32_16x16x32_bf16(
                false, a0.v, false, b0.v, (short)0, acc.v, false, false);
            acc.v = __builtin_amdgcn_wmma_f32_16x16x32_bf16(
                false, a1.v, false, b1.v, (short)0, acc.v, false, false);

            float cn = snorm[j];
            #pragma unroll
            for (int r = 0; r < 8; ++r) {
                float s = cn - 2.f * acc.f[r];
                if (s < minv[r]) { minv[r] = s; mini[r] = j; }  // keeps first min
            }
        }
        // argmin across the 16 lanes of each half (wave32: masks<16 stay in half)
        #pragma unroll
        for (int r = 0; r < 8; ++r) {
            for (int off = 1; off < 16; off <<= 1) {
                float ov = __shfl_xor(minv[r], off, 32);
                int   oi = __shfl_xor(mini[r], off, 32);
                if (ov < minv[r] || (ov == minv[r] && oi < mini[r])) {
                    minv[r] = ov; mini[r] = oi;
                }
            }
        }
        if (m == 0) {
            #pragma unroll
            for (int r = 0; r < 8; ++r)
                idx[tile * 16 + r + 8 * hi] = mini[r];
        }
    }
}

// ---------------------------------------------------------------------------
// Kernel 4: out[b,co,hw] = outTab[idx[p], co]   (straight-through == lookup)
// ---------------------------------------------------------------------------
__global__ void vq_gather_out(const int* __restrict__ idx,
                              const float* __restrict__ outtab,
                              float* __restrict__ out) {
    size_t t = (size_t)blockIdx.x * blockDim.x + threadIdx.x;  // < 16,777,216
    int hw = (int)(t & 1023);
    int co = (int)((t >> 10) & 255);
    int b  = (int)(t >> 18);
    int p  = (b << 10) | hw;
    out[t] = outtab[(size_t)idx[p] * C_OUT + co];
}

// ---------------------------------------------------------------------------
// Kernel 5: loss = (1+beta) * mean((cb[idx] - zp)^2), exact in f32.
// ---------------------------------------------------------------------------
__global__ void vq_loss(const float* __restrict__ zp32,
                        const float* __restrict__ cb,
                        const int* __restrict__ idx,
                        float* __restrict__ loss) {
    __shared__ float wsum[8];
    size_t e = (size_t)blockIdx.x * blockDim.x + threadIdx.x;  // < 4,194,304
    int p = (int)(e >> 6);
    int k = (int)(e & 63);
    int j = idx[p];
    float d = cb[(size_t)j * C_CODE + k] - zp32[e];
    float s = d * d;
    for (int off = 16; off >= 1; off >>= 1) s += __shfl_xor(s, off, 32);
    int lane = threadIdx.x & 31, wave = threadIdx.x >> 5;
    if (lane == 0) wsum[wave] = s;
    __syncthreads();
    if (threadIdx.x == 0) {
        float t = 0.f;
        #pragma unroll
        for (int w = 0; w < 8; ++w) t += wsum[w];
        atomicAdd(loss, t * ((1.f + BETA) / 4194304.f));
    }
}

// ---------------------------------------------------------------------------
// Launch
// ---------------------------------------------------------------------------
extern "C" void kernel_launch(void* const* d_in, const int* in_sizes, int n_in,
                              void* d_out, int out_size, void* d_ws, size_t ws_size,
                              hipStream_t stream) {
    const float* z    = (const float*)d_in[0];
    const float* Win  = (const float*)d_in[1];
    const float* bin  = (const float*)d_in[2];
    const float* cb   = (const float*)d_in[3];
    const float* Wout = (const float*)d_in[4];
    const float* bout = (const float*)d_in[5];

    float* loss = (float*)d_out;          // d_out[0]  = scalar loss
    float* out  = (float*)d_out + 1;      // d_out[1:] = [64,256,32,32] f32

    // workspace carve-up (~27.8 MB total)
    char*   ws     = (char*)d_ws;
    float*  zp32   = (float*)(ws);                          // 16,777,216 B
    __bf16* zp16   = (__bf16*)(ws + 16777216);              //  8,388,608 B
    __bf16* cb16   = (__bf16*)(ws + 25165824);              //    262,144 B
    float*  cbnorm = (float*)(ws + 25427968);               //      8,192 B
    float*  outtab = (float*)(ws + 25436160);               //  2,097,152 B
    int*    idxb   = (int*)(ws + 27533312);                 //    262,144 B

    hipMemsetAsync(loss, 0, sizeof(float), stream);

    vq_prep_cb <<<8,     256, 0, stream>>>(cb, cb16, cbnorm);
    vq_outtab  <<<16,    256, 0, stream>>>(cb, Wout, bout, outtab);
    vq_proj_in <<<512,   256, C_CODE * C_IN * sizeof(float), stream>>>(
        z, Win, bin, zp32, zp16);
    vq_argmin  <<<256,   256, N_CODES * C_CODE * 2 + N_CODES * 4, stream>>>(
        zp16, cb16, cbnorm, idxb);
    vq_gather_out<<<65536, 256, 0, stream>>>(idxb, outtab, out);
    vq_loss    <<<16384, 256, 0, stream>>>(zp32, cb, idxb, loss);
}
